// GRU_4518305595858
// MI455X (gfx1250) — matrix operand order repaired and checked by reading, not compiled, not run on hardware
//
#include <hip/hip_runtime.h>
#include <hip/hip_bf16.h>

// ---------------------------------------------------------------------------
// GRU, T=512, B=64, I=H=512  (MI455X / gfx1250, wave32, WMMA bf16)
//
// Phase 0: convert inputs/weights to bf16, init h ping-pong buffers, zero the
//          global barrier counter.
// Phase 1: GI[32768,1536] = bf16(X) @ bf16(W_ih)^T + bias_ih. Block stages a
//          64x512 B panel in LDS (async global->LDS copy, permuted into WMMA
//          fragment order so reads are conflict-free ds_load_b128), shared by
//          8 waves each computing a 16x64 strip.
// Phase 2: persistent 8-block recurrence, one grid barrier per step; gh slice
//          GEMM with v_wmma_f32_16x16x32_bf16, f32 gate math, h ping-pong.
// ---------------------------------------------------------------------------

typedef __attribute__((ext_vector_type(16))) __bf16 v16bf;
typedef __attribute__((ext_vector_type(8)))  float  v8f;
typedef __attribute__((ext_vector_type(4)))  unsigned int u32x4;
typedef __attribute__((ext_vector_type(4)))  int v4i;

#define T_STEPS 512
#define B_SZ    64
#define I_SZ    512
#define H_SZ    512
#define K3H     1536                    // 3*H
#define M_TOT   (T_STEPS * B_SZ)        // 32768
#define NBLK    8                       // blocks in recurrence kernel

#define AS1 __attribute__((address_space(1)))
#define AS3 __attribute__((address_space(3)))

#if defined(__has_builtin)
# if __has_builtin(__builtin_amdgcn_global_load_async_to_lds_b128) && \
     __has_builtin(__builtin_amdgcn_s_wait_asynccnt)
#  define USE_ASYNC_COPY 1
# endif
#endif
#ifndef USE_ASYNC_COPY
# define USE_ASYNC_COPY 0
#endif

union FragU { u32x4 u[2]; v16bf v; };

// round-to-nearest-even f32 -> bf16
__device__ inline unsigned short f2bf(float f) {
  unsigned int u = __float_as_uint(f);
  u += 0x7FFFu + ((u >> 16) & 1u);
  return (unsigned short)(u >> 16);
}

// A fragment, 16x32 bf16 (MxK), source row-major [rows, ld] of bf16.
// Layout: lane(m=l%16, h=l/16) holds K = h*16 + [0..15] across V0-7.
template <typename P>
__device__ inline v16bf load_a_frag(P base, int ld) {
  int lane = threadIdx.x & 31;
  int m = lane & 15, h = lane >> 4;
  P r = base + (size_t)m * ld;
  FragU f;
  f.u[0] = *(const u32x4*)(r + h * 16);
  f.u[1] = *(const u32x4*)(r + h * 16 + 8);
  return f.v;
}

// B fragment, 32x16 bf16 (KxN), source stored [N, K] row-major.
// Layout (same order as A): lane(n=l%16, h=l/16) holds K = h*16 + [0..15] across V0-7.
template <typename P>
__device__ inline v16bf load_b_frag(P base, int ld) {
  int lane = threadIdx.x & 31;
  int n = lane & 15, h = lane >> 4;
  P r = base + (size_t)n * ld;
  FragU f;
  f.u[0] = *(const u32x4*)(r + h * 16);
  f.u[1] = *(const u32x4*)(r + h * 16 + 8);
  return f.v;
}

// ---------------------------------------------------------------------------
__global__ void f32_to_bf16_kernel(const float* __restrict__ src,
                                   unsigned short* __restrict__ dst, int n) {
  int i = blockIdx.x * 256 + threadIdx.x;
  if (i < n) dst[i] = f2bf(src[i]);
}

__global__ void init_kernel(const float* __restrict__ hidden,
                            float* __restrict__ hF, unsigned short* __restrict__ hB,
                            int* __restrict__ bar) {
  int i = blockIdx.x * 256 + threadIdx.x;
  if (i == 0) *bar = 0;
  if (i < B_SZ * H_SZ) {
    float v = hidden[i];
    hF[i] = v;                 // ping buffer 0 (f32)
    hB[i] = f2bf(v);           // ping buffer 0 (bf16)
  }
}

// ---------------------------------------------------------------------------
// Phase 1: GI = Xbf @ Wihbf^T + bias_ih.
// Grid: 256 m-blocks (128 rows each) x 24 n-groups (64 cols) = 6144 blocks.
// Per block: stage B panel (64 rows x 512 K, 64 KB) in LDS in *fragment
// order*; wave w computes rows [mt*16, mt*16+16), mt = mb*8 + w.
//
// LDS fragment-order layout (16-byte chunks): chunk for (row n, k-octet q):
//   i = n/16 (n-tile), kk = q/4 (k-step), h = (q/2)&1, s = q&1, lane = h*16+n%16
//   addr = (i*16 + kk)*1024 + s*512 + lane*16
// so a wave's fragment read is lds[group + lane*16] and lds[group + 512 + lane*16].
__global__ void __launch_bounds__(256)
gi_gemm_kernel(const unsigned short* __restrict__ Xbf,
               const unsigned short* __restrict__ Wbf,
               const float* __restrict__ bias_ih,
               float* __restrict__ GI) {
  __shared__ char Blds[64 * 1024];    // 64 KB permuted B panel

  const int wave = threadIdx.x >> 5;
  const int lane = threadIdx.x & 31;
  const int mb = blockIdx.x / 24;     // 0..255
  const int ng = blockIdx.x % 24;     // 0..23 (64 cols each)

  // ---- stage W_ih rows [ng*64, ng*64+64) into LDS (permuted) --------------
  {
    const char* gsrc = (const char*)(Wbf + (size_t)ng * 64 * I_SZ);
#pragma unroll
    for (int r = 0; r < 16; ++r) {
      int c = r * 256 + threadIdx.x;            // chunk id 0..4095 (16 B each)
      int n = c >> 6, q = c & 63;               // row, k-octet
      int ti = n >> 4, np = n & 15;
      int kk = q >> 2, h = (q >> 1) & 1, s = q & 1;
      int ldsoff = ((ti * 16 + kk) << 10) + (s << 9) + ((h * 16 + np) << 4);
      const char* gp = gsrc + (size_t)c * 16;   // global chunk is contiguous
#if USE_ASYNC_COPY
      __builtin_amdgcn_global_load_async_to_lds_b128(
          (AS1 v4i*)(v4i*)gp,
          (AS3 v4i*)(v4i*)(Blds + ldsoff), 0, 0);
#else
      *(u32x4*)(Blds + ldsoff) = *(const u32x4*)gp;
#endif
    }
#if USE_ASYNC_COPY
    __builtin_amdgcn_s_wait_asynccnt(0);
#endif
    __syncthreads();
  }

  // ---- 16x64 WMMA strip per wave ------------------------------------------
  const int mt = mb * 8 + wave;                  // 0..2047
  const unsigned short* A0 = Xbf + (size_t)mt * 16 * I_SZ;
  v8f acc[4] = {};
  for (int kk = 0; kk < 16; ++kk) {
    v16bf a = load_a_frag(A0 + kk * 32, I_SZ);
#pragma unroll
    for (int i = 0; i < 4; ++i) {
      const char* bb = Blds + (((i * 16 + kk) << 10) + lane * 16);
      FragU f;
      f.u[0] = *(const u32x4*)bb;
      f.u[1] = *(const u32x4*)(bb + 512);
      acc[i] = __builtin_amdgcn_wmma_f32_16x16x32_bf16(
          false, a, false, f.v, (short)0, acc[i], false, false);
    }
  }

  const int n0 = lane & 15, h = lane >> 4;
#pragma unroll
  for (int i = 0; i < 4; ++i) {
    int col = ng * 64 + i * 16 + n0;
    float bv = bias_ih[col];
#pragma unroll
    for (int r = 0; r < 8; ++r) {
      int row = mt * 16 + r + 8 * h;
      GI[(size_t)row * K3H + col] = acc[i][r] + bv;
    }
  }
}

// ---------------------------------------------------------------------------
// Phase 2: recurrence. 8 blocks x 256 threads (8 waves). Block b owns hidden
// cols j0=64b..j0+63 and W_hh rows {j, H+j, 2H+j}. Wave w: m-tile = w%4,
// n-tile set = w/4 (6 of 12 tiles). One grid barrier per step; h ping-pongs.
__global__ void __launch_bounds__(256)
gru_recur_kernel(const float* __restrict__ GI,
                 const unsigned short* __restrict__ Whhbf,
                 const float* __restrict__ bias_hh,
                 float* __restrict__ hF,           // [2][B*H] f32 ping-pong
                 unsigned short* __restrict__ hB,  // [2][B*H] bf16 ping-pong
                 float* __restrict__ out,          // [T,B,H]
                 float* __restrict__ hfin,         // [B,H]
                 int* __restrict__ bar) {
  __shared__ float ghL[B_SZ * 192];   // 48 KB: gh slice staging
  __shared__ float bhhL[192];

  const int j0   = blockIdx.x * 64;
  const int wave = threadIdx.x >> 5;
  const int lane = threadIdx.x & 31;
  const int mt   = wave & 3;          // m-tile 0..3
  const int nset = wave >> 2;         // 0/1 -> n-tiles nset*6 .. +5

  for (int rr = threadIdx.x; rr < 192; rr += 256) {
    int g = rr / 64, jj = rr % 64;
    bhhL[rr] = bias_hh[g * H_SZ + j0 + jj];
  }
  __syncthreads();

  for (int t = 0; t < T_STEPS; ++t) {
    const int cur = t & 1, nxt = cur ^ 1;
    const unsigned short* hbcur = hB + (size_t)cur * B_SZ * H_SZ;

    // --- gh slice GEMM: [64,512](bf16) x [512,192](bf16) -> f32 ---
    v8f acc[6] = {};
    const unsigned short* A0 = hbcur + (size_t)mt * 16 * H_SZ;
    for (int k = 0; k < H_SZ; k += 32) {
      v16bf a = load_a_frag(A0 + k, H_SZ);
#pragma unroll
      for (int i = 0; i < 6; ++i) {
        int nt = nset * 6 + i;              // 0..11
        int g  = nt >> 2;                   // gate 0..2
        int qc = (nt & 3) * 16;             // col-in-gate base
        const unsigned short* B0 =
            Whhbf + (size_t)(g * H_SZ + j0 + qc) * H_SZ + k;
        v16bf b = load_b_frag(B0, H_SZ);
        acc[i] = __builtin_amdgcn_wmma_f32_16x16x32_bf16(
            false, a, false, b, (short)0, acc[i], false, false);
      }
    }
    // stage gh slice in LDS: local col c = gate*64 + colInGate
    {
      int nloc = lane & 15, h = lane >> 4;
#pragma unroll
      for (int i = 0; i < 6; ++i) {
        int c = (nset * 6 + i) * 16 + nloc;  // == gate*64 + jj
#pragma unroll
        for (int r = 0; r < 8; ++r) {
          int row = mt * 16 + r + 8 * h;
          ghL[row * 192 + c] = acc[i][r];
        }
      }
    }
    __syncthreads();

    // --- gates + h update: 64x64 elements, 16 per thread ---
    const size_t girow = (size_t)t * B_SZ * K3H;
    for (int e = threadIdx.x; e < B_SZ * 64; e += 256) {
      int m = e >> 6, jj = e & 63;
      size_t gidx = girow + (size_t)m * K3H + j0 + jj;
      float ir  = GI[gidx];
      float ii  = GI[gidx + H_SZ];
      float inn = GI[gidx + 2 * H_SZ];
      float hr = ghL[m * 192 + jj]       + bhhL[jj];
      float hi = ghL[m * 192 + 64 + jj]  + bhhL[64 + jj];
      float hn = ghL[m * 192 + 128 + jj] + bhhL[128 + jj];
      float rg = 1.f / (1.f + __expf(-(ir + hr)));
      float ig = 1.f / (1.f + __expf(-(ii + hi)));
      float ng = tanhf(inn + rg * hn);
      int hidx = m * H_SZ + j0 + jj;
      float hp = hF[(size_t)cur * B_SZ * H_SZ + hidx];
      float hy = ng + ig * (hp - ng);
      out[(size_t)(t * B_SZ + m) * H_SZ + j0 + jj] = hy;
      hF[(size_t)nxt * B_SZ * H_SZ + hidx] = hy;
      hB[(size_t)nxt * B_SZ * H_SZ + hidx] = f2bf(hy);
      if (t == T_STEPS - 1) hfin[hidx] = hy;
    }

    // --- grid-wide barrier (8 co-resident blocks, monotonic counter) ---
    __threadfence();
    __syncthreads();
    if (threadIdx.x == 0) {
      __hip_atomic_fetch_add(bar, 1, __ATOMIC_ACQ_REL, __HIP_MEMORY_SCOPE_AGENT);
      const int target = (t + 1) * NBLK;
      while (__hip_atomic_load(bar, __ATOMIC_ACQUIRE, __HIP_MEMORY_SCOPE_AGENT)
             < target) {
        __builtin_amdgcn_s_sleep(1);
      }
    }
    __syncthreads();
  }
}

// ---------------------------------------------------------------------------
extern "C" void kernel_launch(void* const* d_in, const int* in_sizes, int n_in,
                              void* d_out, int out_size, void* d_ws, size_t ws_size,
                              hipStream_t stream) {
  const float* input_    = (const float*)d_in[0];  // [T,B,I]
  const float* hidden    = (const float*)d_in[1];  // [1,B,H]
  const float* weight_ih = (const float*)d_in[2];  // [3H,I]
  const float* weight_hh = (const float*)d_in[3];  // [3H,H]
  const float* bias_ih   = (const float*)d_in[4];  // [3H]
  const float* bias_hh   = (const float*)d_in[5];  // [3H]

  char* ws = (char*)d_ws;
  size_t off = 0;
  unsigned short* Xbf   = (unsigned short*)(ws + off); off += (size_t)M_TOT * I_SZ * 2;   // 32 MB
  unsigned short* Wihbf = (unsigned short*)(ws + off); off += (size_t)K3H * I_SZ * 2;     // 1.5 MB
  unsigned short* Whhbf = (unsigned short*)(ws + off); off += (size_t)K3H * H_SZ * 2;     // 1.5 MB
  float*          GI    = (float*)(ws + off);          off += (size_t)M_TOT * K3H * 4;    // 192 MB
  float*          hF    = (float*)(ws + off);          off += (size_t)2 * B_SZ * H_SZ * 4;
  unsigned short* hB    = (unsigned short*)(ws + off); off += (size_t)2 * B_SZ * H_SZ * 2;
  int*            bar   = (int*)(ws + off);            off += 256;

  float* out  = (float*)d_out;                        // [T,B,H]
  float* hfin = (float*)d_out + (size_t)M_TOT * H_SZ; // [1,B,H]

  // Phase 0: init + bf16 conversion
  init_kernel<<<(B_SZ * H_SZ + 255) / 256, 256, 0, stream>>>(hidden, hF, hB, bar);
  {
    int nX = M_TOT * I_SZ;
    f32_to_bf16_kernel<<<(nX + 255) / 256, 256, 0, stream>>>(input_, Xbf, nX);
    int nW = K3H * I_SZ;
    f32_to_bf16_kernel<<<(nW + 255) / 256, 256, 0, stream>>>(weight_ih, Wihbf, nW);
    f32_to_bf16_kernel<<<(nW + 255) / 256, 256, 0, stream>>>(weight_hh, Whhbf, nW);
  }

  // Phase 1: GI = X @ W_ih^T + b_ih   (256 m-blocks x 24 n-groups)
  gi_gemm_kernel<<<256 * 24, 256, 0, stream>>>(Xbf, Wihbf, bias_ih, GI);

  // Phase 2: sequential recurrence, 8 persistent blocks
  gru_recur_kernel<<<NBLK, 256, 0, stream>>>(GI, Whhbf, bias_hh, hF, hB, out, hfin, bar);
}
